// ObjectPoseEstimator_82463372083408
// MI455X (gfx1250) — matrix operand order repaired and checked by reading, not compiled, not run on hardware
//
#include <hip/hip_runtime.h>
#include <hip/hip_bf16.h>

// ---- static problem sizes (match reference) ----
#define B_GRAPHS 256
#define N_PER    64
#define N_NODES  (B_GRAPHS * N_PER)   // 16384
#define HID      32
#define NF       6
#define OUT_F    2
#define P_PER    (N_PER * (N_PER - 1)) // 4032
#define E_TOTAL  (B_GRAPHS * P_PER)    // 1032192

typedef __attribute__((ext_vector_type(16))) _Float16 v16h;
typedef __attribute__((ext_vector_type(8)))  _Float16 v8h;
typedef __attribute__((ext_vector_type(2)))  _Float16 v2h;
typedef __attribute__((ext_vector_type(8)))  float    v8f;
typedef __attribute__((ext_vector_type(4)))  float    v4f;

__device__ __forceinline__ float fast_sigmoid(float x) {
    return __builtin_amdgcn_rcpf(1.0f + __expf(-x));   // v_rcp_f32, no div-scale
}
__device__ __forceinline__ float fast_tanh(float x) {
    float e = __expf(-2.0f * x);
    return (1.0f - e) * __builtin_amdgcn_rcpf(1.0f + e);
}
// combine two contiguous 16B LDS loads into a 16-half fragment
__device__ __forceinline__ v16h frag2(const _Float16* p0, const _Float16* p1) {
    v8h lo = *(const v8h*)p0;
    v8h hi = *(const v8h*)p1;
    v16h r;
    #pragma unroll
    for (int i = 0; i < 8; ++i) { r[i] = lo[i]; r[8 + i] = hi[i]; }
    return r;
}

// ---------------------------------------------------------------------------
// Kernel A: fold weights on device.
//   w64 = We1 @ We2 (edge MLP collapses: no nonlinearity between its layers)
//   c   = be1 . We2 + be2
//   WzrT[n][k] (f16, TRANSPOSED for contiguous B-fragments):
//       n<32 -> sum_d Wz[d,0,k,n]; n>=32 -> sum_d Wr[d,0,k,n-32]
//   WhT[n][k] (f16, transposed): sum_d Wh[d,0,k,n]
//   bzr = [bz | br]
// ---------------------------------------------------------------------------
__global__ __launch_bounds__(256) void precompute_kernel(
    const float* __restrict__ We1, const float* __restrict__ be1,
    const float* __restrict__ We2, const float* __restrict__ be2,
    const float* __restrict__ Wz,  const float* __restrict__ bz,
    const float* __restrict__ Wr,  const float* __restrict__ br,
    const float* __restrict__ Wh,
    float* __restrict__ w64, float* __restrict__ cval,
    float* __restrict__ bzr, _Float16* __restrict__ WzrT,
    _Float16* __restrict__ WhT)
{
    const int tid = threadIdx.x;
    if (tid < 64) {
        float acc = 0.0f;
        #pragma unroll 8
        for (int j = 0; j < 64; ++j) acc += We1[tid * 64 + j] * We2[j];
        w64[tid] = acc;
        bzr[tid] = (tid < HID) ? bz[tid] : br[tid - HID];
    }
    if (tid == 64) {
        float acc = be2[0];
        #pragma unroll 8
        for (int j = 0; j < 64; ++j) acc += be1[j] * We2[j];
        cval[0] = acc;
    }
    // Wz/Wr/Wh flat layout (2,1,64,32): [d,0,k,o] at d*2048 + k*32 + o
    for (int idx = tid; idx < 64 * 64; idx += 256) {   // idx = n*64 + k
        int n = idx >> 6, k = idx & 63;
        float v = (n < HID)
            ? (Wz[k * HID + n]         + Wz[2048 + k * HID + n])
            : (Wr[k * HID + (n - HID)] + Wr[2048 + k * HID + (n - HID)]);
        WzrT[idx] = (_Float16)v;
    }
    for (int idx = tid; idx < HID * 64; idx += 256) {  // idx = n*64 + k
        int n = idx >> 6, k = idx & 63;
        WhT[idx] = (_Float16)(Wh[k * HID + n] + Wh[2048 + k * HID + n]);
    }
}

// ---------------------------------------------------------------------------
// Kernel B: one block (128 threads = 4 wave32) per graph (64 nodes).
//   x = node_x @ Wn + bn             (VALU, K=6)
//   a_i = x_i . wa ; b_i = x_i . wb  -> workspace (feeds edge kernel)
//   GRU via WMMA f16->f32: ZR = sigmoid([x|h]@Wzr + bzr)
//                          Ht = tanh([x|h*R]@Wh + bh)
//                          H  = Z*h + (1-Z)*Ht
//   out = relu(H) @ Wf + bf
// Each wave owns a 16-node M-tile; K=64 as 2 x K32 WMMA steps. All fragment
// gathers are contiguous 16B LDS loads (A layout is lane-contiguous; weights
// are staged transposed).
// ---------------------------------------------------------------------------
__global__ __launch_bounds__(128) void fused_node_kernel(
    const float* __restrict__ node_x, const float* __restrict__ h_in,
    const float* __restrict__ Wn, const float* __restrict__ bn,
    const float* __restrict__ Wf, const float* __restrict__ bf,
    const float* __restrict__ bh,
    const float* __restrict__ w64, const float* __restrict__ bzr_g,
    const _Float16* __restrict__ WzrT, const _Float16* __restrict__ WhT,
    float* __restrict__ aB, float* __restrict__ bB,
    float* __restrict__ out_head, float* __restrict__ H_out)
{
    __shared__ alignas(16) float    s_x[N_PER * HID];     // 8 KB
    __shared__ alignas(16) float    s_h[N_PER * HID];     // 8 KB
    __shared__ alignas(16) _Float16 s_cat[N_PER * 64];    // 8 KB [x|h] -> [x|h*R]
    __shared__ alignas(16) float    s_zr[N_PER * 64];     // 16 KB
    __shared__ alignas(16) float    s_Hrelu[N_PER * HID]; // 8 KB
    __shared__ alignas(16) _Float16 s_wzrT[64 * 64];      // 8 KB  [n][k]
    __shared__ alignas(16) _Float16 s_whT[HID * 64];      // 4 KB  [n][k]
    __shared__ alignas(16) float    s_bzr[64];

    const int g   = blockIdx.x;
    const int tid = threadIdx.x;

    // ---- stage weights + h with 128-bit copies ----
    {
        const uint4* src = (const uint4*)WzrT;            // 512 x uint4
        uint4* dst = (uint4*)s_wzrT;
        for (int i = tid; i < 512; i += 128) dst[i] = src[i];
        src = (const uint4*)WhT;                          // 256 x uint4
        dst = (uint4*)s_whT;
        for (int i = tid; i < 256; i += 128) dst[i] = src[i];
        src = (const uint4*)(h_in + g * N_PER * HID);     // 512 x uint4
        dst = (uint4*)s_h;
        for (int i = tid; i < 512; i += 128) dst[i] = src[i];
        if (tid < 64) s_bzr[tid] = bzr_g[tid];
    }
    __syncthreads();

    // ---- x = node_x @ Wn + bn ; fill cat = [x|h] (paired f16 stores) ----
    for (int idx = tid; idx < N_PER * (HID / 2); idx += 128) {
        int n = idx >> 4, o = (idx & 15) * 2;
        float a0 = bn[o], a1 = bn[o + 1];
        #pragma unroll
        for (int k = 0; k < NF; ++k) {
            float xv = node_x[(g * N_PER + n) * NF + k];
            a0 += xv * Wn[k * HID + o];
            a1 += xv * Wn[k * HID + o + 1];
        }
        s_x[n * HID + o]     = a0;
        s_x[n * HID + o + 1] = a1;
        v2h px; px[0] = (_Float16)a0; px[1] = (_Float16)a1;
        *(v2h*)&s_cat[n * 64 + o] = px;
        v2h ph; ph[0] = (_Float16)s_h[n * HID + o];
        ph[1] = (_Float16)s_h[n * HID + o + 1];
        *(v2h*)&s_cat[n * 64 + HID + o] = ph;
    }
    __syncthreads();

    // ---- per-node edge scalars: a = x.wa, b = x.wb (vectorized reads) ----
    if (tid < 64) {
        float a = 0.0f, b = 0.0f;
        #pragma unroll
        for (int k = 0; k < HID; k += 4) {
            v4f xv = *(const v4f*)&s_x[tid * HID + k];
            #pragma unroll
            for (int q = 0; q < 4; ++q) {
                a += xv[q] * w64[k + q];
                b += xv[q] * w64[HID + k + q];
            }
        }
        aB[g * N_PER + tid] = a;
        bB[g * N_PER + tid] = b;
    }

    // ---- WMMA tiles (wave32) ----
    const int lane = tid & 31;
    const int wave = tid >> 5;
    const int m0   = wave * 16;
    const int lm   = lane & 15;
    const int lhi  = lane >> 4;       // lanes 0-15 vs 16-31
    const int arow = (m0 + lm) * 64;  // this lane's cat row base

    // A fragments: halves 0-7 <- K = base+lhi*8+(0..7); 8-15 <- K = base+16+lhi*8+(0..7)
    v16h afragX = frag2(&s_cat[arow + lhi * 8], &s_cat[arow + 16 + lhi * 8]);
    v16h afragH = frag2(&s_cat[arow + 32 + lhi * 8], &s_cat[arow + 48 + lhi * 8]);

    // ---- ZR = sigmoid(cat @ Wzr + bzr), 4 N-tiles ----
    #pragma unroll
    for (int nt = 0; nt < 4; ++nt) {
        const int col = nt * 16 + lm;
        const int wb  = col * 64 + lhi * 8;   // transposed weights: [n][k]
        v16h bf0 = frag2(&s_wzrT[wb],      &s_wzrT[wb + 16]);      // K 0..31
        v16h bf1 = frag2(&s_wzrT[wb + 32], &s_wzrT[wb + 48]);      // K 32..63
        v8f acc = {};
        acc = __builtin_amdgcn_wmma_f32_16x16x32_f16(false, afragX, false, bf0,
                                                     (short)0, acc, false, false);
        acc = __builtin_amdgcn_wmma_f32_16x16x32_f16(false, afragH, false, bf1,
                                                     (short)0, acc, false, false);
        float bias = s_bzr[col];
        #pragma unroll
        for (int r = 0; r < 8; ++r) {
            int m = m0 + r + lhi * 8;                  // C/D layout
            s_zr[m * 64 + col] = fast_sigmoid(acc[r] + bias);
        }
    }
    __syncthreads();

    // ---- cat_h = [x | h * R] (paired stores) ----
    for (int idx = tid; idx < N_PER * (HID / 2); idx += 128) {
        int n = idx >> 4, k = (idx & 15) * 2;
        v2h p;
        p[0] = (_Float16)(s_h[n * HID + k]     * s_zr[n * 64 + HID + k]);
        p[1] = (_Float16)(s_h[n * HID + k + 1] * s_zr[n * 64 + HID + k + 1]);
        *(v2h*)&s_cat[n * 64 + HID + k] = p;
    }
    __syncthreads();

    // rebuild h-part A fragment from updated cat
    afragH = frag2(&s_cat[arow + 32 + lhi * 8], &s_cat[arow + 48 + lhi * 8]);

    // ---- H~ = tanh(cat_h @ Wh + bh); H = Z*h + (1-Z)*H~ ----
    #pragma unroll
    for (int nt = 0; nt < 2; ++nt) {
        const int col = nt * 16 + lm;
        const int wb  = col * 64 + lhi * 8;
        v16h bf0 = frag2(&s_whT[wb],      &s_whT[wb + 16]);
        v16h bf1 = frag2(&s_whT[wb + 32], &s_whT[wb + 48]);
        v8f acc = {};
        acc = __builtin_amdgcn_wmma_f32_16x16x32_f16(false, afragX, false, bf0,
                                                     (short)0, acc, false, false);
        acc = __builtin_amdgcn_wmma_f32_16x16x32_f16(false, afragH, false, bf1,
                                                     (short)0, acc, false, false);
        float bias = bh[col];
        #pragma unroll
        for (int r = 0; r < 8; ++r) {
            int   m    = m0 + r + lhi * 8;
            float ht   = fast_tanh(acc[r] + bias);
            float z    = s_zr[m * 64 + col];
            float hold = s_h[m * HID + col];
            float H    = z * hold + (1.0f - z) * ht;
            H_out[(g * N_PER + m) * HID + col] = H;
            s_Hrelu[m * HID + col] = fmaxf(H, 0.0f);
        }
    }
    __syncthreads();

    // ---- out = relu(H) @ Wf + bf : one (node, out) pair per thread ----
    {
        int n = tid >> 1, o = tid & 1;
        float acc = bf[o];
        #pragma unroll
        for (int k = 0; k < HID; k += 4) {
            v4f hv = *(const v4f*)&s_Hrelu[n * HID + k];
            #pragma unroll
            for (int q = 0; q < 4; ++q) acc += hv[q] * Wf[(k + q) * OUT_F + o];
        }
        out_head[(g * N_PER + n) * OUT_F + o] = acc;
    }
}

// ---------------------------------------------------------------------------
// Kernel C: stream out the 1,032,192 edges (memory-bound bulk: ~12.4 MB).
//   edge p within a graph: i = p/63, j = p%63 + (j>=i)   (permutations order)
//   weight = relu(a_i + b_j + c)
// ---------------------------------------------------------------------------
__global__ __launch_bounds__(256) void edge_kernel(
    const float* __restrict__ aB, const float* __restrict__ bB,
    const float* __restrict__ cptr,
    int* __restrict__ idx_out, float* __restrict__ w_out)
{
    const float c = cptr[0];
    const int stride = gridDim.x * blockDim.x;
    for (int e = blockIdx.x * blockDim.x + threadIdx.x; e < E_TOTAL; e += stride) {
        int g = e / P_PER;
        int p = e - g * P_PER;
        int i = p / 63;
        int j = p - i * 63;
        j += (j >= i) ? 1 : 0;
        int ni = g * N_PER + i;
        int nj = g * N_PER + j;
        idx_out[e]           = ni;
        idx_out[E_TOTAL + e] = nj;
        w_out[e] = fmaxf(aB[ni] + bB[nj] + c, 0.0f);
    }
}

// ---------------------------------------------------------------------------
extern "C" void kernel_launch(void* const* d_in, const int* in_sizes, int n_in,
                              void* d_out, int out_size, void* d_ws, size_t ws_size,
                              hipStream_t stream)
{
    const float* node_x = (const float*)d_in[0];
    // d_in[1] = batch_idx (int32) -- unused: graphs are fixed-size blocks of 64
    const float* h   = (const float*)d_in[2];
    const float* Wn  = (const float*)d_in[3];
    const float* bn  = (const float*)d_in[4];
    const float* We1 = (const float*)d_in[5];
    const float* be1 = (const float*)d_in[6];
    const float* We2 = (const float*)d_in[7];
    const float* be2 = (const float*)d_in[8];
    const float* Wz  = (const float*)d_in[9];
    const float* bz  = (const float*)d_in[10];
    const float* Wr  = (const float*)d_in[11];
    const float* br  = (const float*)d_in[12];
    const float* Wh  = (const float*)d_in[13];
    const float* bh  = (const float*)d_in[14];
    const float* Wf  = (const float*)d_in[15];
    const float* bf  = (const float*)d_in[16];

    // workspace layout (~145 KB)
    char* ws = (char*)d_ws;
    float*    w64   = (float*)(ws + 0);                      // 64 f32
    float*    cval  = (float*)(ws + 256);                    // 1 f32
    float*    bzr   = (float*)(ws + 320);                    // 64 f32
    float*    aB    = (float*)(ws + 576);                    // 16384 f32
    float*    bB    = (float*)(ws + 576 + 65536);            // 16384 f32
    _Float16* WzrT  = (_Float16*)(ws + 576 + 131072);        // 64*64 f16 [n][k]
    _Float16* WhT   = (_Float16*)(ws + 576 + 131072 + 8192); // 32*64 f16 [n][k]

    // d_out = [out (16384*2 f32) | H (16384*32 f32) |
    //          edge_index (2*E i32) | edge_weight (E f32)]
    float* out_head = (float*)d_out;
    float* H_out    = (float*)d_out + N_NODES * OUT_F;
    int*   idx_out  = (int*)((float*)d_out + N_NODES * OUT_F + N_NODES * HID);
    float* w_out    = (float*)d_out + N_NODES * OUT_F + N_NODES * HID + 2 * E_TOTAL;

    precompute_kernel<<<1, 256, 0, stream>>>(We1, be1, We2, be2,
                                             Wz, bz, Wr, br, Wh,
                                             w64, cval, bzr, WzrT, WhT);

    fused_node_kernel<<<B_GRAPHS, 128, 0, stream>>>(node_x, h, Wn, bn, Wf, bf, bh,
                                                    w64, bzr, WzrT, WhT,
                                                    aB, bB, out_head, H_out);

    // E_TOTAL = 1008 * 1024 -> exactly 4 coalesced edges per thread
    edge_kernel<<<1008, 256, 0, stream>>>(aB, bB, cval, idx_out, w_out);
}
